// Transformer_4123168604642
// MI455X (gfx1250) — matrix operand order repaired and checked by reading, compiled, not run on hardware
//
#include <hip/hip_runtime.h>

// ---------------------------------------------------------------------------
// MI455X (gfx1250) implementation of the pruning transformer.
// All GEMM-shaped math runs on v_wmma_f32_16x16x32_f16 (wave32, f32 accum).
//
// Fragment layouts used here (C/D per CDNA5 ISA 7.12.2/7.12.4):
//   A operand (16x32, 16-bit): lane = row M (lane&15), K split contiguously
//     by lane half, as B:                  K = e + 16*hi.
//   B operand (32x16, 16-bit): lane = col N (lane&15), K split contiguously
//     by lane half:                        K = j + 16*hi (same as A; not the ISA table order).
//   C/D (16x16 f32): lane = N (lane&15), VGPR r = M = r + 8*hi.
// ---------------------------------------------------------------------------

typedef __attribute__((ext_vector_type(16))) _Float16 v16h;
typedef __attribute__((ext_vector_type(8)))  float    v8f;

#define DEV __device__ __forceinline__

// A-operand fragment: contiguous 16 elements per lane half (K = e + 16*hi).
template <typename TT>
DEV v16h load_fragA(const TT* p, int hi) {
  v16h f;
#pragma unroll
  for (int e = 0; e < 8; ++e) f[e]     = (_Float16)p[16 * hi + e];
#pragma unroll
  for (int e = 0; e < 8; ++e) f[e + 8] = (_Float16)p[16 * hi + 8 + e];
  return f;
}

// B-operand fragment: contiguous 16 elements per lane half (K = j + 16*hi).
template <typename TT>
DEV v16h load_fragB(const TT* p, int hi) {
  v16h f;
#pragma unroll
  for (int j = 0; j < 16; ++j) f[j] = (_Float16)p[16 * hi + j];
  return f;
}

// B-operand gather with strided K (column d of V fixed per lane).
DEV v16h load_fragB_strided(const _Float16* base, long stride, int hi) {
  v16h f;
#pragma unroll
  for (int j = 0; j < 16; ++j) f[j] = base[(long)(16 * hi + j) * stride];
  return f;
}

// ---------------------------------------------------------------------------
// Generic WMMA GEMM:  Out[M,N] = A[M,K] * W[N,K]^T + bias (+ residual)
// One wave computes one 16x16 C tile. arows: optional row indirection for A.
// RESID: 0 none, 1 direct rows of Rsd, 2 gathered rows (rrows) of Rsd.
// ---------------------------------------------------------------------------
template <typename TA, bool OUTH, bool RELU, int RESID>
__global__ void gemm_wmma_ker(const TA* __restrict__ A, int lda,
                              const int* __restrict__ arows,
                              const float* __restrict__ W,
                              const float* __restrict__ bias,
                              const float* __restrict__ Rsd,
                              const int* __restrict__ rrows, int ldr,
                              void* __restrict__ Out, int ldo,
                              int M, int N, int K) {
  const int lane  = threadIdx.x & 31;
  const int wave  = blockIdx.x * (blockDim.x >> 5) + (threadIdx.x >> 5);
  const int tilesN = N >> 4;
  const int tiles  = (M >> 4) * tilesN;
  if (wave >= tiles) return;
  const int tm = wave / tilesN, tn = wave % tilesN;
  const int m = lane & 15, hi = lane >> 4;
  const int ar = tm * 16 + m;
  const TA*    aptr = A + (size_t)(arows ? arows[ar] : ar) * lda;
  const float* wptr = W + (size_t)(tn * 16 + m) * K;
  v8f c = {};
  for (int kb = 0; kb < K; kb += 32) {
    if (kb + 32 < K) __builtin_prefetch(aptr + kb + 32, 0, 1);
    v16h a = load_fragA(aptr + kb, hi);
    v16h b = load_fragB(wptr + kb, hi);
    c = __builtin_amdgcn_wmma_f32_16x16x32_f16(false, a, false, b, (short)0, c,
                                               false, false);
  }
  const int col = tn * 16 + m;
  const float bv = bias ? bias[col] : 0.0f;
#pragma unroll
  for (int r = 0; r < 8; ++r) {
    const int orow = tm * 16 + r + 8 * hi;  // C layout: lane-half selects M+8
    float v = c[r] + bv;
    if (RESID == 1) v += Rsd[(size_t)orow * ldr + col];
    if (RESID == 2) v += Rsd[(size_t)rrows[orow] * ldr + col];
    if (RELU) v = fmaxf(v, 0.0f);
    if (OUTH) ((_Float16*)Out)[(size_t)orow * ldo + col] = (_Float16)v;
    else      ((float*)Out)[(size_t)orow * ldo + col] = v;
  }
}

// In-LDS softmax over 16 rows of length T; 2 lanes per row; optionally folds
// running max-over-heads into maxA.
DEV void softmax16(float* S, float* maxA, int T, int lane) {
  const int row = lane >> 1, half = lane & 1;
  const int c0 = half * (T >> 1), c1 = c0 + (T >> 1);
  float mx = -3.4e38f;
  for (int k = c0; k < c1; ++k) mx = fmaxf(mx, S[row * T + k]);
  mx = fmaxf(mx, __shfl_xor(mx, 1));
  float sum = 0.0f;
  for (int k = c0; k < c1; ++k) {
    float e = __expf(S[row * T + k] - mx);
    S[row * T + k] = e;
    sum += e;
  }
  sum += __shfl_xor(sum, 1);
  const float inv = 1.0f / sum;
  for (int k = c0; k < c1; ++k) {
    float a = S[row * T + k] * inv;
    S[row * T + k] = a;
    if (maxA) maxA[row * T + k] = fmaxf(maxA[row * T + k], a);
  }
}

// Pass 1: per-key importance weight[b,k] = sum_q max_h softmax(QK^T)[b,h,q,k].
// One wave per (b, 16-query tile). S and maxA tiles live in LDS.
__global__ void attn_weight_ker(const _Float16* __restrict__ qkv,
                                float* __restrict__ weight, int T) {
  extern __shared__ float sm[];
  float* S    = sm;
  float* maxA = sm + 16 * T;
  const int lane = threadIdx.x;
  const int nqt = T >> 4;
  const int b  = blockIdx.x / nqt;
  const int qt = blockIdx.x % nqt;
  const int m = lane & 15, hi = lane >> 4;
  const float scale = 0.17677669529663687f;  // 1/sqrt(32)
  for (int i = lane; i < 16 * T; i += 32) maxA[i] = 0.0f;
  __syncthreads();
  const size_t rowQ = (size_t)(b * T + qt * 16 + m) * 768;
  for (int h = 0; h < 8; ++h) {
    v16h a = load_fragA(qkv + rowQ + h * 32, hi);
    for (int kt = 0; kt < nqt; ++kt) {
      const size_t rowK = (size_t)(b * T + kt * 16 + m) * 768 + 256;
      v16h kf = load_fragB(qkv + rowK + h * 32, hi);
      v8f c = {};
      c = __builtin_amdgcn_wmma_f32_16x16x32_f16(false, a, false, kf, (short)0,
                                                 c, false, false);
#pragma unroll
      for (int r = 0; r < 8; ++r)
        S[(r + 8 * hi) * T + kt * 16 + m] = c[r] * scale;
    }
    __syncthreads();
    softmax16(S, maxA, T, lane);
    __syncthreads();
  }
  for (int k = lane; k < T; k += 32) {
    float s = 0.0f;
    for (int r = 0; r < 16; ++r) s += maxA[r * T + k];
    atomicAdd(&weight[b * T + k], s);
  }
}

// Exact jax.lax.top_k replication via rank counting (stable: value desc, idx asc).
__global__ void topk_ker(const float* __restrict__ weight,
                         const int* __restrict__ idx_in,
                         int* __restrict__ inds, int* __restrict__ rowmap,
                         int* __restrict__ idx_out, int T, int L) {
  extern __shared__ float w[];
  const int b = blockIdx.x;
  for (int i = threadIdx.x; i < T; i += blockDim.x) w[i] = weight[b * T + i];
  __syncthreads();
  for (int i = threadIdx.x; i < T; i += blockDim.x) {
    const float wi = w[i];
    int rank = 0;
    for (int j = 0; j < T; ++j)
      rank += (w[j] > wi) || (w[j] == wi && j < i);
    if (rank < L) {
      inds[b * L + rank]    = i;
      rowmap[b * L + rank]  = b * (T + 1) + 1 + i;   // row of rest_g in src
      idx_out[b * L + rank] = idx_in[b * T + i];
    }
  }
}

// Pass 2: recompute softmax rows for the selected queries; O = P*V via WMMA.
// One wave per (b, h, 16-selected-query tile).
__global__ void attn_out_ker(const _Float16* __restrict__ qkv,
                             const int* __restrict__ inds,
                             _Float16* __restrict__ obuf, int T, int L) {
  extern __shared__ float sm[];
  float* S = sm;  // 16*T
  const int lane = threadIdx.x;
  const int nlt = L >> 4;
  const int b  = blockIdx.x / (8 * nlt);
  const int h  = (blockIdx.x / nlt) % 8;
  const int lt = blockIdx.x % nlt;
  const int m = lane & 15, hi = lane >> 4;
  const float scale = 0.17677669529663687f;
  const int q = inds[b * L + lt * 16 + m];
  v16h a = load_fragA(qkv + (size_t)(b * T + q) * 768 + h * 32, hi);
  const int nkt = T >> 4;
  for (int kt = 0; kt < nkt; ++kt) {
    const size_t rowK = (size_t)(b * T + kt * 16 + m) * 768 + 256;
    v16h kf = load_fragB(qkv + rowK + h * 32, hi);
    v8f c = {};
    c = __builtin_amdgcn_wmma_f32_16x16x32_f16(false, a, false, kf, (short)0, c,
                                               false, false);
#pragma unroll
    for (int r = 0; r < 8; ++r)
      S[(r + 8 * hi) * T + kt * 16 + m] = c[r] * scale;
  }
  __syncthreads();
  softmax16(S, nullptr, T, lane);
  __syncthreads();
#pragma unroll
  for (int nh = 0; nh < 2; ++nh) {  // dh=32 -> two 16-wide N halves
    const int d = h * 32 + nh * 16 + m;
    v8f c = {};
    for (int kc = 0; kc < T; kc += 32) {
      v16h p  = load_fragA(S + (size_t)m * T + kc, hi);
      v16h vf = load_fragB_strided(qkv + (size_t)(b * T + kc) * 768 + 512 + d,
                                   768, hi);
      c = __builtin_amdgcn_wmma_f32_16x16x32_f16(false, p, false, vf, (short)0,
                                                 c, false, false);
    }
#pragma unroll
    for (int r = 0; r < 8; ++r) {
      const int orow = lt * 16 + r + 8 * hi;
      obuf[(size_t)(b * L + orow) * 256 + h * 32 + nh * 16 + m] = (_Float16)c[r];
    }
  }
}

// Single-query (token) MHA core: one wave per (b,h).
__global__ void summary_attn_ker(const _Float16* __restrict__ qtok,
                                 const _Float16* __restrict__ ks,
                                 const _Float16* __restrict__ vs,
                                 _Float16* __restrict__ attnout, int L) {
  extern __shared__ float sc[];
  const int lane = threadIdx.x;
  const int b = blockIdx.x >> 3;
  const int h = blockIdx.x & 7;
  const float scale = 0.17677669529663687f;
  for (int l = lane; l < L; l += 32) {
    float s = 0.0f;
    const _Float16* kp = ks + (size_t)(b * L + l) * 256 + h * 32;
    const _Float16* qp = qtok + (size_t)b * 256 + h * 32;
    for (int d = 0; d < 32; ++d) s += (float)qp[d] * (float)kp[d];
    sc[l] = s * scale;
  }
  __syncthreads();
  float mx = -3.4e38f;
  for (int l = lane; l < L; l += 32) mx = fmaxf(mx, sc[l]);
  for (int o = 16; o; o >>= 1) mx = fmaxf(mx, __shfl_xor(mx, o));
  float sum = 0.0f;
  for (int l = lane; l < L; l += 32) {
    float e = __expf(sc[l] - mx);
    sc[l] = e;
    sum += e;
  }
  for (int o = 16; o; o >>= 1) sum += __shfl_xor(sum, o);
  const float inv = 1.0f / sum;
  __syncthreads();
  float o = 0.0f;
  for (int l = 0; l < L; ++l)
    o += sc[l] * (float)vs[(size_t)(b * L + l) * 256 + h * 32 + lane];
  attnout[(size_t)b * 256 + h * 32 + lane] = (_Float16)(o * inv);
}

// LayerNorm over D=256, one block per row; mode 1 also scatters into the next
// src buffer at row b*(L+1)+1+l.
__global__ void layernorm_ker(const float* __restrict__ X,
                              const float* __restrict__ w,
                              const float* __restrict__ bias,
                              float* __restrict__ out1,
                              float* __restrict__ out2, int mode, int Lpar) {
  __shared__ float p1[8], p2[8], stats[2];
  const int r = blockIdx.x;
  const int t = threadIdx.x;
  const float x = X[(size_t)r * 256 + t];
  float s = x, q = x * x;
  for (int o = 16; o; o >>= 1) { s += __shfl_xor(s, o); q += __shfl_xor(q, o); }
  if ((t & 31) == 0) { p1[t >> 5] = s; p2[t >> 5] = q; }
  __syncthreads();
  if (t == 0) {
    float a = 0.0f, b2 = 0.0f;
    for (int i = 0; i < 8; ++i) { a += p1[i]; b2 += p2[i]; }
    const float mean = a * (1.0f / 256.0f);
    const float var  = b2 * (1.0f / 256.0f) - mean * mean;
    stats[0] = mean;
    stats[1] = rsqrtf(var + 1e-5f);
  }
  __syncthreads();
  const float y = (x - stats[0]) * stats[1] * w[t] + bias[t];
  out1[(size_t)r * 256 + t] = y;
  if (mode == 1) out2[(size_t)(r + r / Lpar + 1) * 256 + t] = y;
}

__global__ void fill_maps_ker(int* restmap, int* tokmap, int T) {
  const int r = blockIdx.x * blockDim.x + threadIdx.x;
  if (r < 128 * T) restmap[r] = (r / T) * (T + 1) + 1 + (r % T);
  if (r < 128) tokmap[r] = r * (T + 1);
}

__global__ void zero_ker(float* p, int n) {
  const int i = blockIdx.x * blockDim.x + threadIdx.x;
  if (i < n) p[i] = 0.0f;
}

// token -> src_next row 0 and d_out token slot with (g,bb) swizzle.
__global__ void scatter_token_ker(const float* __restrict__ tok,
                                  float* __restrict__ src_next,
                                  float* __restrict__ out_tok, int L) {
  const int i = blockIdx.x * blockDim.x + threadIdx.x;  // 128*256
  const int b = i >> 8, d = i & 255;
  const float v = tok[i];
  src_next[(size_t)b * (L + 1) * 256 + d] = v;
  out_tok[(size_t)((b & 31) * 4 + (b >> 5)) * 256 + d] = v;
}

__global__ void idxout_ker(const int* __restrict__ idx, float* __restrict__ out,
                           int n) {
  const int i = blockIdx.x * blockDim.x + threadIdx.x;
  if (i < n) out[i] = (float)idx[i];
}

// ---------------------------------------------------------------------------
extern "C" void kernel_launch(void* const* d_in, const int* in_sizes, int n_in,
                              void* d_out, int out_size, void* d_ws,
                              size_t ws_size, hipStream_t stream) {
  (void)in_sizes; (void)n_in; (void)out_size; (void)ws_size;
  const float* in_src     = (const float*)d_in[0];
  const int*   in_idx     = (const int*)d_in[1];
  const float* mix_in_w   = (const float*)d_in[2];
  const float* mix_in_b   = (const float*)d_in[3];
  const float* mix_out_w  = (const float*)d_in[4];
  const float* mix_out_b  = (const float*)d_in[5];
  const float* sdb_norm_w = (const float*)d_in[6];
  const float* sdb_norm_b = (const float*)d_in[7];
  const float* sdb_nc_w   = (const float*)d_in[8];
  const float* sdb_nc_b   = (const float*)d_in[9];
  const float* ffn1_w     = (const float*)d_in[10];
  const float* ffn1_b     = (const float*)d_in[11];
  const float* ffn2_w     = (const float*)d_in[12];
  const float* ffn2_b     = (const float*)d_in[13];
  const float* sa_in_w    = (const float*)d_in[14];
  const float* sa_in_b    = (const float*)d_in[15];
  const float* sa_out_w   = (const float*)d_in[16];
  const float* sa_out_b   = (const float*)d_in[17];
  const float* lin1_w     = (const float*)d_in[18];
  const float* lin1_b     = (const float*)d_in[19];
  const float* lin2_w     = (const float*)d_in[20];
  const float* lin2_b     = (const float*)d_in[21];
  const float* norm1_w    = (const float*)d_in[22];
  const float* norm1_b    = (const float*)d_in[23];
  const float* norm2_w    = (const float*)d_in[24];
  const float* norm2_b    = (const float*)d_in[25];

  char* w8 = (char*)d_ws;
  size_t off = 0;
  auto alloc = [&](size_t bytes) -> void* {
    void* p = w8 + off;
    off = (off + bytes + 255) & ~(size_t)255;
    return p;
  };
  float*    srcA    = (float*)alloc(128ull * 385 * 256 * 4);
  float*    srcB    = (float*)alloc(128ull * 385 * 256 * 4);
  _Float16* qkv     = (_Float16*)alloc(128ull * 384 * 768 * 2);
  float*    weight  = (float*)alloc(128ull * 384 * 4);
  int*      inds    = (int*)alloc(128ull * 192 * 4);
  int*      rowmap  = (int*)alloc(128ull * 192 * 4);
  int*      restmap = (int*)alloc(128ull * 384 * 4);
  int*      tokmap  = (int*)alloc(128ull * 4);
  int*      idxA    = (int*)alloc(128ull * 384 * 4);
  int*      idxB    = (int*)alloc(128ull * 384 * 4);
  _Float16* obuf    = (_Float16*)alloc(128ull * 192 * 256 * 2);
  float*    xbuf    = (float*)alloc(128ull * 192 * 256 * 4);
  float*    xnorm   = (float*)alloc(128ull * 192 * 256 * 4);
  float*    xm2     = (float*)alloc(128ull * 192 * 256 * 4);
  float*    sintra  = (float*)alloc(128ull * 192 * 256 * 4);
  _Float16* h1      = (_Float16*)alloc(128ull * 192 * 512 * 2);
  _Float16* tokq    = (_Float16*)alloc(128ull * 256 * 2);
  _Float16* sak     = (_Float16*)alloc(128ull * 192 * 256 * 2);
  _Float16* sav     = (_Float16*)alloc(128ull * 192 * 256 * 2);
  _Float16* aout    = (_Float16*)alloc(128ull * 256 * 2);
  float*    sum1    = (float*)alloc(128ull * 256 * 4);
  float*    tokln   = (float*)alloc(128ull * 256 * 4);
  _Float16* h2      = (_Float16*)alloc(128ull * 1024 * 2);
  float*    t2b     = (float*)alloc(128ull * 256 * 4);
  float*    tok2    = (float*)alloc(128ull * 256 * 4);

  hipMemcpyAsync(srcA, in_src, 128ull * 385 * 256 * 4,
                 hipMemcpyDeviceToDevice, stream);
  hipMemcpyAsync(idxA, in_idx, 128ull * 384 * 4, hipMemcpyDeviceToDevice,
                 stream);

  float* outp = (float*)d_out;
  const size_t TOK_BASE = 128ull * 49 * 256;       // 1,605,632
  const size_t IDX_BASE = TOK_BASE + 3ull * 32768; // 1,703,936

  const int Ts[3] = {384, 192, 96};
  float* scur = srcA; float* snxt = srcB;
  int* icur = idxA;   int* inxt = idxB;

  for (int i = 0; i < 3; ++i) {
    const int T = Ts[i], L = T / 2;
    const int BT = 128 * T, BL = 128 * L;

    fill_maps_ker<<<(BT + 255) / 256, 256, 0, stream>>>(restmap, tokmap, T);
    zero_ker<<<(BT + 255) / 256, 256, 0, stream>>>(weight, BT);

    {  // QKV = rest @ mix_in_w^T + b  (f16 out)
      int waves = (BT / 16) * (768 / 16);
      gemm_wmma_ker<float, true, false, 0><<<(waves + 3) / 4, 128, 0, stream>>>(
          scur, 256, restmap, mix_in_w + (size_t)i * 768 * 256,
          mix_in_b + i * 768, nullptr, nullptr, 0, qkv, 768, BT, 768, 256);
    }
    attn_weight_ker<<<128 * (T / 16), 32, 2 * 16 * T * 4, stream>>>(qkv, weight, T);
    topk_ker<<<128, 256, T * 4, stream>>>(weight, icur, inds, rowmap, inxt, T, L);
    attn_out_ker<<<128 * 8 * (L / 16), 32, 16 * T * 4, stream>>>(qkv, inds, obuf, T, L);
    {  // x = rest_g + O @ mix_out_w^T + b
      int waves = (BL / 16) * (256 / 16);
      gemm_wmma_ker<_Float16, false, false, 2><<<(waves + 3) / 4, 128, 0, stream>>>(
          obuf, 256, nullptr, mix_out_w + (size_t)i * 256 * 256,
          mix_out_b + i * 256, scur, rowmap, 256, xbuf, 256, BL, 256, 256);
    }
    layernorm_ker<<<BL, 256, 0, stream>>>(xbuf, sdb_norm_w + i * 256,
                                          sdb_norm_b + i * 256, xnorm, nullptr, 0, 0);
    {  // h = relu(xnorm @ ffn1^T + b)
      int waves = (BL / 16) * (512 / 16);
      gemm_wmma_ker<float, true, true, 0><<<(waves + 3) / 4, 128, 0, stream>>>(
          xnorm, 256, nullptr, ffn1_w + (size_t)i * 512 * 256, ffn1_b + i * 512,
          nullptr, nullptr, 0, h1, 512, BL, 512, 256);
    }
    {  // xm2 = xnorm + h @ ffn2^T + b
      int waves = (BL / 16) * (256 / 16);
      gemm_wmma_ker<_Float16, false, false, 1><<<(waves + 3) / 4, 128, 0, stream>>>(
          h1, 512, nullptr, ffn2_w + (size_t)i * 256 * 512, ffn2_b + i * 256,
          xnorm, nullptr, 256, xm2, 256, BL, 256, 512);
    }
    // src_intra = LN(xm2); also scattered into src_next rows 1..L
    layernorm_ker<<<BL, 256, 0, stream>>>(xm2, sdb_nc_w + i * 256,
                                          sdb_nc_b + i * 256, sintra, snxt, 1, L);
    {  // token q projection (rows gathered from src row 0 per batch)
      int waves = (128 / 16) * (256 / 16);
      gemm_wmma_ker<float, true, false, 0><<<(waves + 3) / 4, 128, 0, stream>>>(
          scur, 256, tokmap, sa_in_w + (size_t)i * 768 * 256, sa_in_b + i * 768,
          nullptr, nullptr, 0, tokq, 256, 128, 256, 256);
    }
    {  // k / v projections of src_intra
      int waves = (BL / 16) * (256 / 16);
      gemm_wmma_ker<float, true, false, 0><<<(waves + 3) / 4, 128, 0, stream>>>(
          sintra, 256, nullptr, sa_in_w + (size_t)i * 768 * 256 + 256 * 256,
          sa_in_b + i * 768 + 256, nullptr, nullptr, 0, sak, 256, BL, 256, 256);
      gemm_wmma_ker<float, true, false, 0><<<(waves + 3) / 4, 128, 0, stream>>>(
          sintra, 256, nullptr, sa_in_w + (size_t)i * 768 * 256 + 512 * 256,
          sa_in_b + i * 768 + 512, nullptr, nullptr, 0, sav, 256, BL, 256, 256);
    }
    summary_attn_ker<<<128 * 8, 32, L * 4, stream>>>(tokq, sak, sav, aout, L);
    {  // sum1 = token + attn @ sa_out^T + b
      int waves = (128 / 16) * (256 / 16);
      gemm_wmma_ker<_Float16, false, false, 2><<<(waves + 3) / 4, 128, 0, stream>>>(
          aout, 256, nullptr, sa_out_w + (size_t)i * 256 * 256,
          sa_out_b + i * 256, scur, tokmap, 256, sum1, 256, 128, 256, 256);
    }
    layernorm_ker<<<128, 256, 0, stream>>>(sum1, norm1_w + i * 256,
                                           norm1_b + i * 256, tokln, nullptr, 0, 0);
    {  // h2 = relu(tokln @ lin1^T + b)
      int waves = (128 / 16) * (1024 / 16);
      gemm_wmma_ker<float, true, true, 0><<<(waves + 3) / 4, 128, 0, stream>>>(
          tokln, 256, nullptr, lin1_w + (size_t)i * 1024 * 256,
          lin1_b + i * 1024, nullptr, nullptr, 0, h2, 1024, 128, 1024, 256);
    }
    {  // t2 = tokln + h2 @ lin2^T + b
      int waves = (128 / 16) * (256 / 16);
      gemm_wmma_ker<_Float16, false, false, 1><<<(waves + 3) / 4, 128, 0, stream>>>(
          h2, 1024, nullptr, lin2_w + (size_t)i * 256 * 1024, lin2_b + i * 256,
          tokln, nullptr, 256, t2b, 256, 128, 256, 1024);
    }
    layernorm_ker<<<128, 256, 0, stream>>>(t2b, norm2_w + i * 256,
                                           norm2_b + i * 256, tok2, nullptr, 0, 0);
    scatter_token_ker<<<128, 256, 0, stream>>>(tok2, snxt,
                                               outp + TOK_BASE + (size_t)i * 32768, L);
    { float* t = scur; scur = snxt; snxt = t; }
    { int* t = icur; icur = inxt; inxt = t; }
  }

  // Final outputs: src [128,49,256], then tokens (written above), then indices.
  hipMemcpyAsync(outp, scur, 128ull * 49 * 256 * 4, hipMemcpyDeviceToDevice,
                 stream);
  idxout_ker<<<(6144 + 255) / 256, 256, 0, stream>>>(icur, outp + IDX_BASE, 6144);
}
